// TensorizedSyntaxOnlyCCG_51376398795523
// MI455X (gfx1250) — matrix operand order repaired and checked by reading, not compiled, not run on hardware
//
#include <hip/hip_runtime.h>

// TensorizedSyntaxOnlyCCG — CKY inside algorithm via WMMA f16 GEMM on gfx1250.
//
// spans[s][i](c) = logsumexp_{m,a,b} spans[m][i](a) + spans[s-m][m+i](b) + T[a,b,c]
// as exp-domain GEMM: rows=(split,cell,batch), K=9216 (a,b), N=96 (c):
//   X[row, a*96+b] = exp(L[a]-lmax)*exp(R[b]-rmax)      (fp16, built on the fly)
//   expT[a*96+b,c] = exp(T[a,b,c]-tmax[c])              (fp16, pre-packed B-fragment order)
//   out = log(X @ expT) + lmax + rmax + tmax[c], then logsumexp over m.
// K split into 8 segments writing disjoint f32 partial slices (deterministic).
// Inner loop: manual 2x-unrolled ping-pong pipeline (a0/b0 vs a1/b1) so the
// register allocator keeps accumulators fixed — no acc moves, partial waits only.

typedef _Float16 v16h __attribute__((ext_vector_type(16)));
typedef _Float16 h2   __attribute__((ext_vector_type(2)));
typedef float    v8f  __attribute__((ext_vector_type(8)));

#define NT   96
#define NB   32
#define SEQ  16
#define SPAN_STRIDE 3072          // NB*NT
#define KCHUNKS 288               // 9216 / 32
#define KSEG 8
#define SEGCH (KCHUNKS / KSEG)    // 36 (even)
#define MAXROWS 2048              // max (s-1)*(17-s)*32 over s
#define PACKED_ELEMS (KCHUNKS * 6 * 32 * 16)   // 884736 fp16

__device__ __forceinline__ size_t spanOff(int s, int i) {
  return (size_t)(((s - 1) * SEQ + i) * SPAN_STRIDE);
}

// ---------------- prep: tmax[c] = max_{a,b} T[a,b,c] ----------------
__global__ void ccg_tmax(const float* __restrict__ T, float* __restrict__ tmax) {
  __shared__ float red[256];
  const int c = blockIdx.x;
  float mx = -INFINITY;
  for (int ab = threadIdx.x; ab < NT * NT; ab += 256)
    mx = fmaxf(mx, T[(size_t)ab * NT + c]);
  red[threadIdx.x] = mx;
  __syncthreads();
  for (int off = 128; off > 0; off >>= 1) {
    if (threadIdx.x < off)
      red[threadIdx.x] = fmaxf(red[threadIdx.x], red[threadIdx.x + off]);
    __syncthreads();
  }
  if (threadIdx.x == 0) tmax[c] = red[0];
}

// ---------------- prep: pack exp(T - tmax) into WMMA B-fragment order --------
// Element (kc, nt, lane, h): N = nt*16 + lane%16 ; K = 16*(lane/16) + h
//   global k = kc*32 + K  ->  a = kc/3 , b = (kc%3)*32 + K
__global__ void ccg_pack(const float* __restrict__ T, const float* __restrict__ tmax,
                         _Float16* __restrict__ expTp) {
  int idx = blockIdx.x * 256 + threadIdx.x;
  if (idx >= PACKED_ELEMS) return;
  const int h  = idx & 15;
  const int L  = (idx >> 4) & 31;
  const int fr = idx >> 9;            // kc*6 + nt
  const int nt = fr % 6;
  const int kc = fr / 6;
  const int a  = kc / 3;
  const int b  = (kc - a * 3) * 32 + (L >> 4) * 16 + h;
  const int c  = nt * 16 + (L & 15);
  expTp[idx] = (_Float16)__expf(T[((size_t)a * NT + b) * NT + c] - tmax[c]);
}

// ---------------- prep: spans[1][i][b][c] = dist[b][i][c] ----------------
__global__ void ccg_init(const float* __restrict__ dist, float* __restrict__ spanbuf) {
  int idx = blockIdx.x * 256 + threadIdx.x;
  if (idx >= SEQ * NB * NT) return;
  const int c = idx % NT;
  const int r = idx / NT;
  const int b = r % NB;
  const int i = r / NB;
  spanbuf[spanOff(1, i) + (size_t)b * NT + c] = dist[((size_t)b * SEQ + i) * NT + c];
}

// ---------------- main WMMA GEMM per diagonal ----------------
struct BFrags { v16h b[6]; };
struct ALoad  { _Float16 ea; h2 e[8]; };

__device__ __forceinline__ BFrags loadB(const _Float16* __restrict__ expTp,
                                        int kc, int lane) {
  BFrags r;
  const _Float16* bbase = expTp + (((size_t)kc * 192 + lane) << 4);
  r.b[0] = *(const v16h*)(bbase);
  r.b[1] = *(const v16h*)(bbase + 512);
  r.b[2] = *(const v16h*)(bbase + 1024);
  r.b[3] = *(const v16h*)(bbase + 1536);
  r.b[4] = *(const v16h*)(bbase + 2048);
  r.b[5] = *(const v16h*)(bbase + 2560);
  return r;
}

__device__ __forceinline__ ALoad loadA(const _Float16 (*__restrict__ eaS)[100],
                                       const _Float16 (*__restrict__ ebS)[100],
                                       int kc, int rowA, int g2) {
  ALoad r;
  const int a  = kc / 3;
  const int bb = (kc - a * 3) << 5;
  r.ea = eaS[rowA][a];
  const h2* p0 = (const h2*)&ebS[rowA][bb + g2 * 16];
  const h2* p1 = (const h2*)&ebS[rowA][bb + g2 * 16 + 8];
  r.e[0] = p0[0]; r.e[1] = p0[1]; r.e[2] = p0[2]; r.e[3] = p0[3];
  r.e[4] = p1[0]; r.e[5] = p1[1]; r.e[6] = p1[2]; r.e[7] = p1[3];
  return r;
}

__device__ __forceinline__ v16h mulA(const ALoad& L) {
  h2 av; av[0] = L.ea; av[1] = L.ea;
  union { v16h v; h2 p[8]; } af;
  af.p[0] = av * L.e[0]; af.p[1] = av * L.e[1];
  af.p[2] = av * L.e[2]; af.p[3] = av * L.e[3];
  af.p[4] = av * L.e[4]; af.p[5] = av * L.e[5];
  af.p[6] = av * L.e[6]; af.p[7] = av * L.e[7];
  return af.v;
}

#define WMMA6(A, B)                                                                                   \
  do {                                                                                                \
    acc0 = __builtin_amdgcn_wmma_f32_16x16x32_f16(false, (A), false, (B).b[0], (short)0, acc0, false, false); \
    acc1 = __builtin_amdgcn_wmma_f32_16x16x32_f16(false, (A), false, (B).b[1], (short)0, acc1, false, false); \
    acc2 = __builtin_amdgcn_wmma_f32_16x16x32_f16(false, (A), false, (B).b[2], (short)0, acc2, false, false); \
    acc3 = __builtin_amdgcn_wmma_f32_16x16x32_f16(false, (A), false, (B).b[3], (short)0, acc3, false, false); \
    acc4 = __builtin_amdgcn_wmma_f32_16x16x32_f16(false, (A), false, (B).b[4], (short)0, acc4, false, false); \
    acc5 = __builtin_amdgcn_wmma_f32_16x16x32_f16(false, (A), false, (B).b[5], (short)0, acc5, false, false); \
  } while (0)

// grid.x = tiles * KSEG ; block = 128 (4 waves, one 16-row M-tile each).
// Rows: g = ((m * w) + i)*32 + batch ; tile covers 64 rows.
__global__ __launch_bounds__(128) void ccg_gemm(
    const float* __restrict__ spanbuf, const _Float16* __restrict__ expTp,
    float* __restrict__ lrsumG, float* __restrict__ scoresExp,
    int s, int w, int totalRows) {
  __shared__ _Float16 eaS[64][100];   // stride 100 halves (50 dwords) breaks bank conflicts
  __shared__ _Float16 ebS[64][100];
  __shared__ float    lrS[64];

  const int tile   = blockIdx.x >> 3;   // / KSEG
  const int seg    = blockIdx.x & (KSEG - 1);
  const int wgBase = tile * 64;
  const int t      = threadIdx.x;

  // ---- stage exp(L - lmax), exp(R - rmax) into LDS (threads<64: L, >=64: R) ----
  {
    const int rowLocal = t & 63;
    const bool isR = t >= 64;
    const int g = wgBase + rowLocal;
    const bool valid = g < totalRows;
    const float* cell = nullptr;
    if (valid) {
      const int m   = g / (w * NB);            // split index 0..s-2
      const int rem = g - m * (w * NB);
      const int i   = rem / NB;
      const int b   = rem - i * NB;
      const int ss  = isR ? (s - (m + 1)) : (m + 1);
      const int pos = isR ? ((m + 1) + i) : i;
      cell = spanbuf + spanOff(ss, pos) + (size_t)b * NT;
    }
    float mx = -INFINITY;
    if (valid)
      for (int c = 0; c < NT; ++c) mx = fmaxf(mx, cell[c]);
    if (!isR) lrS[rowLocal] = valid ? mx : 0.0f;
    __syncthreads();
    if (isR && valid) lrS[rowLocal] += mx;
    _Float16* dst = isR ? &ebS[rowLocal][0] : &eaS[rowLocal][0];
    if (valid)
      for (int c = 0; c < NT; ++c) dst[c] = (_Float16)__expf(cell[c] - mx);
    else
      for (int c = 0; c < NT; ++c) dst[c] = (_Float16)0.0f;
    __syncthreads();
    if (!isR && valid && seg == 0) lrsumG[g] = lrS[rowLocal];
  }

  // ---- ping-pong pipelined WMMA loop (no loop-carried copies) ----
  const int lane = t & 31;
  const int wv   = t >> 5;
  const int rowA = wv * 16 + (lane & 15);     // A layout: M = lane%16
  const int g2   = lane >> 4;                 // selects K half-range

  v8f acc0 = {}, acc1 = {}, acc2 = {}, acc3 = {}, acc4 = {}, acc5 = {};
  const int kc0 = seg * SEGCH;

  v16h   a0 = mulA(loadA(eaS, ebS, kc0, rowA, g2));
  BFrags b0 = loadB(expTp, kc0, lane);

#pragma unroll 1
  for (int i = 0; i < SEGCH - 2; i += 2) {
    ALoad  al1 = loadA(eaS, ebS, kc0 + i + 1, rowA, g2);   // DScnt prefetch
    BFrags b1  = loadB(expTp, kc0 + i + 1, lane);          // LOADcnt prefetch
    WMMA6(a0, b0);
    v16h a1 = mulA(al1);

    ALoad al2 = loadA(eaS, ebS, kc0 + i + 2, rowA, g2);
    b0 = loadB(expTp, kc0 + i + 2, lane);                  // refill slot 0
    WMMA6(a1, b1);
    a0 = mulA(al2);
  }
  // epilogue: chunks kc0+34 (in a0/b0) and kc0+35
  {
    ALoad  alL = loadA(eaS, ebS, kc0 + SEGCH - 1, rowA, g2);
    BFrags bL  = loadB(expTp, kc0 + SEGCH - 1, lane);
    WMMA6(a0, b0);
    v16h aL = mulA(alL);
    WMMA6(aL, bL);
  }

  // ---- store exp-domain partials (D layout: M = r + 8*(lane/16), N = lane%16) ----
  const int cb = lane & 15;
  float* outSeg = scoresExp + (size_t)seg * (MAXROWS * NT);
  for (int r = 0; r < 8; ++r) {
    const int localRow = wv * 16 + r + 8 * g2;
    const int g = wgBase + localRow;
    if (g < totalRows) {
      float* orow = outSeg + (size_t)g * NT;
      orow[cb]      = acc0[r];
      orow[16 + cb] = acc1[r];
      orow[32 + cb] = acc2[r];
      orow[48 + cb] = acc3[r];
      orow[64 + cb] = acc4[r];
      orow[80 + cb] = acc5[r];
    }
  }
}

// ---------------- per diagonal: log + rescale + logsumexp over splits --------
__global__ void ccg_reduce(const float* __restrict__ scoresExp,
                           const float* __restrict__ lrsumG,
                           const float* __restrict__ tmax,
                           float* __restrict__ spanbuf, int s, int w) {
  int idx = blockIdx.x * 256 + threadIdx.x;
  const int total = w * NB * NT;
  if (idx >= total) return;
  const int c   = idx % NT;
  const int rem = idx / NT;
  const int b   = rem % NB;
  const int i   = rem / NB;
  const float tm = tmax[c];
  const int nm = s - 1;
  float vm[SEQ - 1];
  float mx = -INFINITY;
  for (int m = 0; m < nm; ++m) {
    const int row = (m * w + i) * NB + b;
    float E = 0.0f;
    for (int seg = 0; seg < KSEG; ++seg)
      E += scoresExp[(size_t)seg * (MAXROWS * NT) + (size_t)row * NT + c];
    const float v = __logf(E) + lrsumG[row] + tm;
    vm[m] = v;
    mx = fmaxf(mx, v);
  }
  float out;
  if (mx == -INFINITY) {
    out = mx;
  } else {
    float sum = 0.0f;
    for (int m = 0; m < nm; ++m) sum += __expf(vm[m] - mx);
    out = mx + __logf(sum);
  }
  spanbuf[spanOff(s, i) + (size_t)b * NT + c] = out;
}

// ---------------- final gather ----------------
__global__ void ccg_final(const int* __restrict__ wl, const int* __restrict__ root,
                          const float* __restrict__ spanbuf, float* __restrict__ out) {
  const int b = threadIdx.x;
  if (b >= NB) return;
  const int s = wl[b];                       // length in [1, 16]
  out[b] = spanbuf[spanOff(s, 0) + (size_t)b * NT + root[0]];
}

extern "C" void kernel_launch(void* const* d_in, const int* in_sizes, int n_in,
                              void* d_out, int out_size, void* d_ws, size_t ws_size,
                              hipStream_t stream) {
  (void)in_sizes; (void)n_in; (void)out_size; (void)ws_size;
  // inputs: 0=words (unused), 1=words_length i32[32], 2=dist f32[32,16,96],
  //         3=T f32[96,96,96], 4=allowed_root_type_indices i32[1]
  const int*   wl   = (const int*)d_in[1];
  const float* dist = (const float*)d_in[2];
  const float* T    = (const float*)d_in[3];
  const int*   root = (const int*)d_in[4];

  char* ws = (char*)d_ws;
  float*    tmax      = (float*)(ws + 0);              //    384 B
  _Float16* expTp     = (_Float16*)(ws + 512);         //  1.77 MB
  float*    spanbuf   = (float*)(ws + 1769984);        //  3.15 MB
  float*    lrsumG    = (float*)(ws + 4915712);        //   8 KB
  float*    scoresExp = (float*)(ws + 4923904);        //  6.29 MB (8 seg slices)

  ccg_tmax<<<NT, 256, 0, stream>>>(T, tmax);
  ccg_pack<<<(PACKED_ELEMS + 255) / 256, 256, 0, stream>>>(T, tmax, expTp);
  ccg_init<<<(SEQ * NB * NT + 255) / 256, 256, 0, stream>>>(dist, spanbuf);

  for (int s = 2; s <= SEQ; ++s) {
    const int w     = SEQ - s + 1;
    const int rows  = (s - 1) * w * NB;
    const int tiles = (rows + 63) / 64;
    ccg_gemm<<<tiles * KSEG, 128, 0, stream>>>(spanbuf, expTp, lrsumG, scoresExp,
                                               s, w, rows);
    const int tot = w * NB * NT;
    ccg_reduce<<<(tot + 255) / 256, 256, 0, stream>>>(scoresExp, lrsumG, tmax,
                                                      spanbuf, s, w);
  }
  ccg_final<<<1, 32, 0, stream>>>(wl, root, spanbuf, (float*)d_out);
}
